// SphericalConv2d_11579231830339
// MI455X (gfx1250) — matrix operand order repaired and checked by reading, not compiled, not run on hardware
//
#include <hip/hip_runtime.h>
#include <hip/hip_bf16.h>
#include <math.h>

// ---------------------------------------------------------------------------
// SphericalConv2d on gfx1250 (CDNA5, wave32, WMMA f32_16x16x32_f16)
//
// Pipeline (all stages as f16 WMMA GEMMs with f32 accumulate):
//   A: XF  = Xh[65536x512] * Ft[512x256]          (rfft, m<128, re|im)
//   B: Cc  = Pana_m[256x256] * XFt_(ri,m)[256x256] (Legendre analysis, 256 planes)
//   C: Gc  = complex channel mix per (l,m) mode    (WMMA 16x16x32, K=CIN)
//   D: Gh  = PsynT_m[256x256] * Gc_(ri,m)[256x256] (Legendre synthesis, 256 planes)
//   E: out = Gh[65536x256] * Fi[256x512]           (irfft)
//
// GEMM staging uses CDNA5 GLOBAL_LOAD_ASYNC_TO_LDS_B128 (ASYNCcnt) with a
// double-buffered LDS pipeline: tile i+1 streams in while tile i is in WMMA.
// ---------------------------------------------------------------------------

typedef __attribute__((ext_vector_type(16))) _Float16 v16h;
typedef __attribute__((ext_vector_type(8)))  float    v8f;

#define NB    8
#define NC    32
#define NBC   256      // NB*NC
#define NJ    256      // nlat
#define NN    512      // nlon
#define NL    256      // lmax
#define NM    128      // retained m modes
#define NRIM  256      // re|im concat of m
#define NR    65536    // NBC*NJ rows
#define TWO_PI_D 6.283185307179586476925286766559

// ---- workspace layout (bytes) ----------------------------------------------
static constexpr size_t OFF_FT  = 0;                                   // [256][512] f16
static constexpr size_t OFF_FI  = OFF_FT  + (size_t)NRIM*NN*2;         // [512][256] f16
static constexpr size_t OFF_PAN = OFF_FI  + (size_t)NN*NRIM*2;         // [128][256(l)][256(j)] f16
static constexpr size_t OFF_PSY = OFF_PAN + (size_t)NM*NL*NJ*2;        // [128][256(j)][256(l)] f16
static constexpr size_t OFF_XH  = OFF_PSY + (size_t)NM*NJ*NL*2;        // [65536][512] f16
static constexpr size_t OFF_XFT = OFF_XH  + (size_t)NR*NN*2;           // [256(rim)][65536] f16
static constexpr size_t OFF_CC  = OFF_XFT + (size_t)NRIM*NR*2;         // [256(rim)][256(l)][256(bc)] f32
static constexpr size_t OFF_WB  = OFF_CC  + (size_t)NRIM*NL*NBC*4;     // [256(l)][128(m)][2][32(o)][32(i)] f16
static constexpr size_t OFF_GC  = OFF_WB  + (size_t)NL*NM*2*NC*NC*2;   // [256(rim)][256(bo)][256(l)] f16
static constexpr size_t OFF_GH  = OFF_GC  + (size_t)NRIM*NBC*NL*2;     // [65536(bo,j)][256(rim)] f16

// ---- CDNA5 async memory->LDS helpers ---------------------------------------
__device__ __forceinline__ void async_b128(unsigned lds_off, unsigned long long gaddr) {
    // GLOBAL_LOAD_ASYNC_TO_LDS_B128: per-lane 16B memory -> LDS, ASYNCcnt-tracked
    asm volatile("global_load_async_to_lds_b128 %0, %1, off"
                 :: "v"(lds_off), "v"(gaddr) : "memory");
}
__device__ __forceinline__ void wait_async4() {
    asm volatile("s_wait_asynccnt 0x4" ::: "memory");   // oldest tile done, next in flight
}
__device__ __forceinline__ void wait_async0() {
    asm volatile("s_wait_asynccnt 0x0" ::: "memory");
}
__device__ __forceinline__ unsigned lds_off_of(const void* p) {
    // flat shared address low 32 bits == LDS byte offset (aperture calc)
    return (unsigned)(unsigned long long)(uintptr_t)p;
}

// ---------------------------------------------------------------------------
// K0: DFT basis tables (f16, fragment-friendly layouts)
//   Ft[col][n]: col<128 -> cos(2pi*col*n/512)*s ; col>=128 -> -sin(...)*s
//   Fi[n][mm] : mm<128  -> c_m*cos(...)/sqrt(2pi); mm>=128 -> -c_m*sin(...)
// ---------------------------------------------------------------------------
__global__ void k_trig(_Float16* __restrict__ Ft, _Float16* __restrict__ Fi) {
    int idx = blockIdx.x * 256 + threadIdx.x;
    const double s_ana = sqrt(TWO_PI_D) / (double)NN;
    const double s_syn = 1.0 / sqrt(TWO_PI_D);
    if (idx < NRIM * NN) {
        int col = idx >> 9, n = idx & (NN - 1);
        int mm = col & (NM - 1);
        double ang = (TWO_PI_D * (double)mm * (double)n) / (double)NN;
        double v = ((col < NM) ? cos(ang) : -sin(ang)) * s_ana;
        Ft[(size_t)col * NN + n] = (_Float16)v;
    } else {
        idx -= NRIM * NN;
        int n = idx >> 8, c = idx & (NRIM - 1);
        int mm = c & (NM - 1);
        double coef = (mm == 0) ? ((c < NM) ? 1.0 : 0.0) : 2.0;  // Im(G0) ignored by irfft
        double ang = (TWO_PI_D * (double)mm * (double)n) / (double)NN;
        double v = ((c < NM) ? cos(ang) : -sin(ang)) * coef * s_syn;
        Fi[(size_t)n * NRIM + c] = (_Float16)v;
    }
}

// ---------------------------------------------------------------------------
// K1: associated Legendre tables via the reference recurrence (f64 on device)
//   Pana[m][l][j] = Pbar_l^m(cos th_j) * w_j   (analysis, A row-major per m)
//   PsynT[m][j][l] = Pbar_l^m(cos th_j)        (synthesis, A row-major per m)
// ---------------------------------------------------------------------------
__global__ void k_legendre(_Float16* __restrict__ Pana, _Float16* __restrict__ PsynT) {
    const int m = blockIdx.x;       // 0..127
    const int j = threadIdx.x;      // 0..255
    const double theta = M_PI * ((double)j + 0.5) / (double)NJ;
    const double ct = cos(theta), st = sin(theta);
    const double wq = st * M_PI / (double)NJ;
    _Float16* pa = Pana  + (size_t)m * NL * NJ;
    _Float16* ps = PsynT + (size_t)m * NJ * NL + (size_t)j * NL;
    for (int l = 0; l < m; ++l) { pa[(size_t)l * NJ + j] = (_Float16)0.0f; ps[l] = (_Float16)0.0f; }
    double pmm = 1.0 / sqrt(2.0);
    for (int mm = 1; mm <= m; ++mm) pmm *= sqrt((2.0 * mm + 1.0) / (2.0 * mm)) * st;
    pa[(size_t)m * NJ + j] = (_Float16)(pmm * wq);
    ps[m] = (_Float16)pmm;
    if (m + 1 < NL) {
        double plm1 = sqrt(2.0 * m + 3.0) * ct * pmm;   // P[m+1][m]
        double plm2 = pmm;
        pa[(size_t)(m + 1) * NJ + j] = (_Float16)(plm1 * wq);
        ps[m + 1] = (_Float16)plm1;
        for (int l = m + 2; l < NL; ++l) {
            double ll = (double)l * (double)l;
            double a = sqrt((4.0 * ll - 1.0) / (ll - (double)m * m));
            double b = sqrt(((2.0 * l + 1.0) * (double)(l - 1 - m) * (double)(l - 1 + m)) /
                            ((2.0 * l - 3.0) * (ll - (double)m * m)));
            double p = a * ct * plm1 - b * plm2;
            plm2 = plm1; plm1 = p;
            pa[(size_t)l * NJ + j] = (_Float16)(p * wq);
            ps[l] = (_Float16)p;
        }
    }
}

// ---------------------------------------------------------------------------
// K2: x (f32) -> Xh (f16), identical [row=(b*32+c)*256+j][n] layout
// ---------------------------------------------------------------------------
__global__ void k_xconv(const float* __restrict__ x, _Float16* __restrict__ Xh) {
    size_t idx = (size_t)blockIdx.x * 256 + threadIdx.x;
    Xh[idx] = (_Float16)x[idx];
}

// ---------------------------------------------------------------------------
// K3: weights f32 [i][o][lw][m] -> f16 WB[l][m][w(r|i)][o][i]  (B-frag layout)
// ---------------------------------------------------------------------------
__global__ void k_wconv(const float* __restrict__ w1r, const float* __restrict__ w1i,
                        const float* __restrict__ w2r, const float* __restrict__ w2i,
                        _Float16* __restrict__ WB) {
    __shared__ _Float16 sh[2][8][1024];   // 32 KB
    const int l  = blockIdx.x >> 4;
    const int m0 = (blockIdx.x & 15) * 8;
    const int lw = l & (NM - 1);
    const float* wr = (l < NM) ? w1r : w2r;
    const float* wi = (l < NM) ? w1i : w2i;
    for (int idx = threadIdx.x; idx < 2048; idx += 256) {
        int w = idx >> 10, io = idx & 1023;       // io = i*32 + o
        int i = io >> 5, o = io & 31;
        const float* src = (w ? wi : wr) + ((size_t)io * NM + lw) * NM + m0;
        float4 v0 = *(const float4*)(src);
        float4 v1 = *(const float4*)(src + 4);
        int d = o * 32 + i;                       // [o][i] for B fragment
        sh[w][0][d] = (_Float16)v0.x; sh[w][1][d] = (_Float16)v0.y;
        sh[w][2][d] = (_Float16)v0.z; sh[w][3][d] = (_Float16)v0.w;
        sh[w][4][d] = (_Float16)v1.x; sh[w][5][d] = (_Float16)v1.y;
        sh[w][6][d] = (_Float16)v1.z; sh[w][7][d] = (_Float16)v1.w;
    }
    __syncthreads();
    for (int idx = threadIdx.x; idx < 2048; idx += 256) {
        int w = idx >> 10, rem = idx & 1023;
        int mm = rem >> 7, seg = rem & 127;
        *(uint4*)(WB + (((size_t)l * NM + m0 + mm) * 2 + w) * 1024 + seg * 8) =
            *(const uint4*)&sh[w][mm][seg * 8];
    }
}

// ---------------------------------------------------------------------------
// Generic 128x128-tile WMMA GEMM core. 256 threads = 8 waves; wave = 32x64.
// A: row-major f16 (K contiguous), B: column-major f16 (K contiguous per col).
// Fragment layouts per CDNA5 ISA 7.12.2:
//   A 16x32:  lane m=l&15, hi=l>>4; halves[0..7]=K(hi?8..15:0..7),
//                                   halves[8..15]=K(hi?24..31:16..23)
//   B 32x16:  lane col=l&15, hi=l>>4; same K order as A
// Double-buffered LDS fed by global_load_async_to_lds_b128 (ASYNCcnt).
// ---------------------------------------------------------------------------
template <typename Writer>
__device__ __forceinline__ void gemm_core(const _Float16* __restrict__ Ag, int lda,
                                          const _Float16* __restrict__ Bg, int ldb,
                                          int K, int m0, int n0, Writer wr) {
    __shared__ _Float16 As[2][128 * 40];   // row stride 40 halves (pad vs bank conflicts)
    __shared__ _Float16 Bs[2][128 * 40];
    const int tid  = threadIdx.x;
    const int lane = tid & 31, wid = tid >> 5;
    const int mw = (wid >> 1) * 32, nw = (wid & 1) * 64;
    const int fm = lane & 15, hi = lane >> 4;

    // staging coords: thread handles 16B chunks (r0,c0) and (r0+64,c0) of A and B
    const int r0 = tid >> 2, c0 = tid & 3;
    const _Float16* gA0 = Ag + (size_t)(m0 + r0) * lda + c0 * 8;
    const _Float16* gA1 = gA0 + (size_t)64 * lda;
    const _Float16* gB0 = Bg + (size_t)(n0 + r0) * ldb + c0 * 8;
    const _Float16* gB1 = gB0 + (size_t)64 * ldb;
    const int soff = r0 * 40 + c0 * 8;
    unsigned lA0[2], lA1[2], lB0[2], lB1[2];
#pragma unroll
    for (int b = 0; b < 2; ++b) {
        lA0[b] = lds_off_of(&As[b][soff]);
        lA1[b] = lds_off_of(&As[b][soff + 64 * 40]);
        lB0[b] = lds_off_of(&Bs[b][soff]);
        lB1[b] = lds_off_of(&Bs[b][soff + 64 * 40]);
    }
    auto issue = [&](int k0, int b) {
        async_b128(lA0[b], (unsigned long long)(uintptr_t)(gA0 + k0));
        async_b128(lA1[b], (unsigned long long)(uintptr_t)(gA1 + k0));
        async_b128(lB0[b], (unsigned long long)(uintptr_t)(gB0 + k0));
        async_b128(lB1[b], (unsigned long long)(uintptr_t)(gB1 + k0));
    };

    v8f zero = {};
    v8f acc[2][4];
#pragma unroll
    for (int i = 0; i < 2; ++i)
#pragma unroll
        for (int j = 0; j < 4; ++j) acc[i][j] = zero;

    const int steps = K >> 5;
    issue(0, 0);
    for (int s = 0; s < steps; ++s) {
        const int buf = s & 1;
        if (s + 1 < steps) {
            issue((s + 1) * 32, buf ^ 1);   // stream next tile while computing this one
            wait_async4();                  // oldest 4 (this tile) landed in LDS
        } else {
            wait_async0();
        }
        __syncthreads();

        v16h af[2], bf[4];
#pragma unroll
        for (int i = 0; i < 2; ++i) {
            int base = (mw + i * 16 + fm) * 40 + hi * 8;
            ((uint4*)&af[i])[0] = *(const uint4*)&As[buf][base];
            ((uint4*)&af[i])[1] = *(const uint4*)&As[buf][base + 16];
        }
#pragma unroll
        for (int j = 0; j < 4; ++j) {
            int base = (nw + j * 16 + fm) * 40 + hi * 8;
            ((uint4*)&bf[j])[0] = *(const uint4*)&Bs[buf][base];
            ((uint4*)&bf[j])[1] = *(const uint4*)&Bs[buf][base + 16];
        }
#pragma unroll
        for (int i = 0; i < 2; ++i)
#pragma unroll
            for (int j = 0; j < 4; ++j)
                acc[i][j] = __builtin_amdgcn_wmma_f32_16x16x32_f16(
                    false, af[i], false, bf[j], (short)0, acc[i][j], false, false);
        __syncthreads();   // all waves done reading buf before it is refilled
    }
#pragma unroll
    for (int i = 0; i < 2; ++i)
#pragma unroll
        for (int j = 0; j < 4; ++j) {
            int row0 = m0 + mw + i * 16 + hi * 8;
            int col  = n0 + nw + j * 16 + fm;
#pragma unroll
            for (int r = 0; r < 8; ++r) wr(row0 + r, col, acc[i][j][r]);
        }
}

// ---- stage A: Xh[65536x512] * Ft[512x256] -> XFt16[rim][row] ---------------
__global__ void k_gemmA(const _Float16* __restrict__ Xh, const _Float16* __restrict__ Ft,
                        _Float16* __restrict__ XFt) {
    int m0 = blockIdx.x * 128, n0 = blockIdx.y * 128;
    gemm_core(Xh, NN, Ft, NN, NN, m0, n0,
              [=](int row, int col, float v) {
                  XFt[(size_t)col * NR + row] = (_Float16)v;
              });
}

// ---- stage B: per (ri,m): Pana_m[256x256(l,j)] * XFt[256x256(bc,j)] --------
__global__ void k_gemmB(const _Float16* __restrict__ Pana, const _Float16* __restrict__ XFt,
                        float* __restrict__ Cc) {
    int z = blockIdx.z;                               // rim plane
    const _Float16* A = Pana + (size_t)(z & (NM - 1)) * NL * NJ;
    const _Float16* Bm = XFt + (size_t)z * NR;
    float* out = Cc + (size_t)z * NL * NBC;
    gemm_core(A, NJ, Bm, NJ, NJ, blockIdx.x * 128, blockIdx.y * 128,
              [=](int row, int col, float v) { out[(size_t)row * NBC + col] = v; });
}

// ---- stage C: complex channel mix, one (l,m) mode per wave -----------------
// A frag rows: 0-7 = xr[b], 8-15 = xi[b]; K = i (CIN). Two WMMAs (Wr,Wi).
__global__ void __launch_bounds__(128)
k_mix(const float* __restrict__ Cc, const _Float16* __restrict__ WB,
      _Float16* __restrict__ Gc) {
    const int wid = threadIdx.x >> 5, lane = threadIdx.x & 31;
    const int gid = blockIdx.x * 4 + wid;             // 0..32767
    const int l = gid >> 7, m = gid & (NM - 1);
    const int rowp = lane & 15, hi = lane >> 4;
    const int ri = rowp >> 3, b = rowp & 7;

    const float* src = Cc + (size_t)(ri * NM + m) * NL * NBC + (size_t)l * NBC + b * NC;
    float tf[16];
    *(float4*)&tf[0]  = *(const float4*)(src + hi * 8);
    *(float4*)&tf[4]  = *(const float4*)(src + hi * 8 + 4);
    *(float4*)&tf[8]  = *(const float4*)(src + 16 + hi * 8);
    *(float4*)&tf[12] = *(const float4*)(src + 16 + hi * 8 + 4);
    v16h a;
#pragma unroll
    for (int t = 0; t < 16; ++t) a[t] = (_Float16)tf[t];

    const _Float16* wb = WB + ((size_t)l * NM + m) * 2048;
    v8f zero = {};
    v8f accR[2], accI[2];
#pragma unroll
    for (int och = 0; och < 2; ++och) {
        int o = och * 16 + rowp;  // lane column = output channel
        v16h br, bi;
        ((uint4*)&br)[0] = *(const uint4*)(wb + (size_t)o * 32 + hi * 8);
        ((uint4*)&br)[1] = *(const uint4*)(wb + (size_t)o * 32 + hi * 8 + 16);
        ((uint4*)&bi)[0] = *(const uint4*)(wb + 1024 + (size_t)o * 32 + hi * 8);
        ((uint4*)&bi)[1] = *(const uint4*)(wb + 1024 + (size_t)o * 32 + hi * 8 + 16);
        accR[och] = __builtin_amdgcn_wmma_f32_16x16x32_f16(false, a, false, br, (short)0, zero, false, false);
        accI[och] = __builtin_amdgcn_wmma_f32_16x16x32_f16(false, a, false, bi, (short)0, zero, false, false);
    }
    // lanes 0-15 hold rows b=r (xr*W); lanes 16-31 hold rows b=r (xi*W)
    const size_t baseRe = ((size_t)m * NBC) * NL + (size_t)l;
    const size_t baseIm = ((size_t)(NM + m) * NBC) * NL + (size_t)l;
#pragma unroll
    for (int och = 0; och < 2; ++och)
#pragma unroll
        for (int r = 0; r < 8; ++r) {
            float re = accR[och][r] - __shfl_xor(accI[och][r], 16);
            float im = accI[och][r] + __shfl_xor(accR[och][r], 16);
            if (lane < 16) {
                int bo = r * NC + och * 16 + lane;
                Gc[baseRe + (size_t)bo * NL] = (_Float16)re;
                Gc[baseIm + (size_t)bo * NL] = (_Float16)im;
            }
        }
}

// ---- stage D: per (ri,m): PsynT_m[256x256(j,l)] * Gc[256x256(bo,l)] --------
__global__ void k_gemmD(const _Float16* __restrict__ PsynT, const _Float16* __restrict__ Gc,
                        _Float16* __restrict__ Gh) {
    int z = blockIdx.z;
    const _Float16* A = PsynT + (size_t)(z & (NM - 1)) * NJ * NL;
    const _Float16* Bm = Gc + (size_t)z * NBC * NL;
    gemm_core(A, NL, Bm, NL, NL, blockIdx.x * 128, blockIdx.y * 128,
              [=](int row, int col, float v) {   // row=j, col=bo
                  Gh[((size_t)col * NJ + row) * NRIM + z] = (_Float16)v;
              });
}

// ---- stage E: Gh[65536x256] * Fi[256x512] -> d_out f32 ---------------------
__global__ void k_gemmE(const _Float16* __restrict__ Gh, const _Float16* __restrict__ Fi,
                        float* __restrict__ out) {
    int m0 = blockIdx.x * 128, n0 = blockIdx.y * 128;
    gemm_core(Gh, NRIM, Fi, NRIM, NRIM, m0, n0,
              [=](int row, int col, float v) { out[(size_t)row * NN + col] = v; });
}

// ---------------------------------------------------------------------------
extern "C" void kernel_launch(void* const* d_in, const int* in_sizes, int n_in,
                              void* d_out, int out_size, void* d_ws, size_t ws_size,
                              hipStream_t stream) {
    const float* x   = (const float*)d_in[0];
    const float* w1r = (const float*)d_in[1];
    const float* w1i = (const float*)d_in[2];
    const float* w2r = (const float*)d_in[3];
    const float* w2i = (const float*)d_in[4];
    float* out = (float*)d_out;

    char* ws = (char*)d_ws;
    _Float16* Ft  = (_Float16*)(ws + OFF_FT);
    _Float16* Fi  = (_Float16*)(ws + OFF_FI);
    _Float16* Pan = (_Float16*)(ws + OFF_PAN);
    _Float16* Psy = (_Float16*)(ws + OFF_PSY);
    _Float16* Xh  = (_Float16*)(ws + OFF_XH);
    _Float16* XFt = (_Float16*)(ws + OFF_XFT);
    float*    Cc  = (float*)   (ws + OFF_CC);
    _Float16* WB  = (_Float16*)(ws + OFF_WB);
    _Float16* Gc  = (_Float16*)(ws + OFF_GC);
    _Float16* Gh  = (_Float16*)(ws + OFF_GH);

    k_trig    <<<1024, 256, 0, stream>>>(Ft, Fi);
    k_legendre<<<128, 256, 0, stream>>>(Pan, Psy);
    k_xconv   <<<131072, 256, 0, stream>>>(x, Xh);
    k_wconv   <<<4096, 256, 0, stream>>>(w1r, w1i, w2r, w2i, WB);
    k_gemmA   <<<dim3(512, 2, 1), 256, 0, stream>>>(Xh, Ft, XFt);
    k_gemmB   <<<dim3(2, 2, 256), 256, 0, stream>>>(Pan, XFt, Cc);
    k_mix     <<<8192, 128, 0, stream>>>(Cc, WB, Gc);
    k_gemmD   <<<dim3(2, 2, 256), 256, 0, stream>>>(Psy, Gc, Gh);
    k_gemmE   <<<dim3(512, 4, 1), 256, 0, stream>>>(Gh, Fi, out);
}
